// AFlowNet_40690520162855
// MI455X (gfx1250) — matrix operand order repaired and checked by reading, not compiled, not run on hardware
//
#include <hip/hip_runtime.h>

#define LEAK 0.1f

typedef __bf16 bf16_t;
typedef bf16_t v16bf __attribute__((ext_vector_type(16)));
typedef bf16_t v8bf  __attribute__((ext_vector_type(8)));
typedef float  v8f   __attribute__((ext_vector_type(8)));

__device__ __forceinline__ bf16_t f2bf(float f) {
    unsigned u = __builtin_bit_cast(unsigned, f);
    u += 0x7fffu + ((u >> 16) & 1u);                 // round-to-nearest-even
    unsigned short h = (unsigned short)(u >> 16);
    return __builtin_bit_cast(bf16_t, h);
}

__device__ __forceinline__ float leaky(float v) { return v > 0.0f ? v : LEAK * v; }

// two 16-byte contiguous LDS chunks -> one 16-half fragment (2x ds_load_b128)
__device__ __forceinline__ v16bf ld_frag(const bf16_t* p0, const bf16_t* p1) {
    v8bf a = *(const v8bf*)p0;
    v8bf b = *(const v8bf*)p1;
    return __builtin_shufflevector(a, b, 0, 1, 2, 3, 4, 5, 6, 7,
                                         8, 9, 10, 11, 12, 13, 14, 15);
}

// ---------------------------------------------------------------------------
// Implicit-GEMM 3x3 conv (pad=1, stride=1) via V_WMMA_F32_16X16X32_BF16.
// M = H*W pixels, N = Cout (all NT=ceil(Cout/16) tiles handled in-block so the
// A tile is staged once per K-step and reused), K = Cin*9.
// 4 waves/block; wave w owns M-subtile (blockIdx.x*64 + w*16).
// A tiles per-wave, B tile (all N) shared; both staged so each lane's
// fragments are contiguous 16-byte LDS runs (ds_load_b128). K-step = 64.
// ---------------------------------------------------------------------------
template <int NT>
__global__ __launch_bounds__(128) void k_conv3x3_wmma(
    const float* __restrict__ in0, int Cin0,
    const float* __restrict__ in1, int Cin1,
    const float* __restrict__ wgt,   // [Cout, Cin*9] (OIHW flattened)
    const float* __restrict__ bias,  // [Cout]
    float* __restrict__ out,         // [B, Cout, H, W]
    int Cout, int H, int W, int do_leaky)
{
    const int Cin = Cin0 + Cin1;
    const int K   = Cin * 9;
    const int HW  = H * W;

    const int lane  = threadIdx.x & 31;
    const int wv    = threadIdx.x >> 5;
    const int m0    = blockIdx.x * 64 + wv * 16;   // first output pixel of this wave
    const int b     = blockIdx.z;

    // row stride 72 halves = 144 B (16-byte aligned rows and chunk offsets)
    __shared__ bf16_t lA[4][16][72];       // per-wave A tile: [m][k], k in [0,64)
    __shared__ bf16_t lB[NT * 16][72];     // shared B tile, transposed: [n][k]

    const int mrow  = lane & 15;
    const int khalf = lane >> 4;           // 0: lanes 0-15, 1: lanes 16-31

    // hoisted per-pixel coordinates for this wave's 16 rows
    int yv[16], yWv[16], xv[16];
    bool pv[16];
    #pragma unroll
    for (int t = 0; t < 16; ++t) {
        const int p = m0 + t;
        pv[t]  = (p < HW);
        const int y = p / W;
        yv[t]  = y;
        yWv[t] = y * W;
        xv[t]  = p - y * W;
    }

    v8f acc[NT];
    #pragma unroll
    for (int nt = 0; nt < NT; ++nt) acc[nt] = {};

    for (int k0 = 0; k0 < K; k0 += 64) {
        // ---- stage A: 16x64 per wave; lane fills columns kk=lane and lane+32
        #pragma unroll
        for (int c = 0; c < 2; ++c) {
            const int kk = lane + 32 * c;
            const int k  = k0 + kk;
            const bool kvalid = (k < K);
            int rW = 0, r = 0, s = 0;
            const float* src = in0;
            if (kvalid) {
                const int ci = k / 9;
                const int rs = k - ci * 9;
                const int rr = rs / 3;
                r  = rr - 1;
                s  = (rs - rr * 3) - 1;
                rW = r * W;
                src = (ci < Cin0)
                    ? in0 + (size_t)(b * Cin0 + ci) * HW
                    : in1 + (size_t)(b * Cin1 + (ci - Cin0)) * HW;
            }
            #pragma unroll
            for (int t = 0; t < 16; ++t) {
                float val = 0.0f;
                const int yy = yv[t] + r;
                const int xx = xv[t] + s;
                if (kvalid && pv[t] &&
                    (unsigned)yy < (unsigned)H && (unsigned)xx < (unsigned)W) {
                    val = src[yWv[t] + rW + xx];
                }
                lA[wv][t][kk] = f2bf(val);
            }
        }
        // ---- stage B (transposed) for ALL N tiles: NT*16*64 elems / 128 thr
        #pragma unroll
        for (int t = 0; t < NT * 8; ++t) {
            const int q  = threadIdx.x + 128 * t;
            const int n  = q >> 6;         // 0 .. NT*16-1
            const int kk = q & 63;
            float val = 0.0f;
            if ((k0 + kk) < K && n < Cout)
                val = wgt[n * K + (k0 + kk)];
            lB[n][kk] = f2bf(val);
        }
        __syncthreads();

        // ---- two K=32 chunks; A fragment reused across all N tiles
        #pragma unroll
        for (int c = 0; c < 2; ++c) {
            const bf16_t* ap = &lA[wv][mrow][32 * c + khalf * 8];
            const v16bf af   = ld_frag(ap, ap + 16);
            #pragma unroll
            for (int nt = 0; nt < NT; ++nt) {
                const bf16_t* bp = &lB[nt * 16 + mrow][32 * c + khalf * 8];
                const v16bf bfv  = ld_frag(bp, bp + 16);
                acc[nt] = __builtin_amdgcn_wmma_f32_16x16x32_bf16(
                              false, af, false, bfv, (short)0, acc[nt],
                              false, false);
            }
        }
        __syncthreads();
    }

    // ---- epilogue: C/D layout -> VGPR i holds (M = i + 8*khalf, N = lane%16)
    #pragma unroll
    for (int nt = 0; nt < NT; ++nt) {
        const int n = nt * 16 + mrow;
        #pragma unroll
        for (int i = 0; i < 8; ++i) {
            const int p = m0 + i + khalf * 8;
            if (p < HW && n < Cout) {
                float v = acc[nt][i] + bias[n];
                if (do_leaky) v = leaky(v);
                out[(b * Cout + n) * HW + p] = v;
            }
        }
    }
}

// ---------------------------------------------------------------------------
// 7x7 correlation (mean over C) + fused leaky ReLU.  out: [B,49,H,W]
// ---------------------------------------------------------------------------
__global__ void k_corr7(const float* __restrict__ a, const float* __restrict__ c,
                        float* __restrict__ out, int B, int C, int H, int W)
{
    const int HW = H * W;
    const int idx = blockIdx.x * blockDim.x + threadIdx.x;
    const int total = B * 49 * HW;
    if (idx >= total) return;
    const int p  = idx % HW;
    const int ch = (idx / HW) % 49;
    const int b  = idx / (49 * HW);
    const int y  = p / W, x = p % W;
    const int dy = ch / 7 - 3, dx = ch % 7 - 3;
    const int yy = y + dy, xx = x + dx;
    float acc = 0.0f;
    if (yy >= 0 && yy < H && xx >= 0 && xx < W) {
        const float* pa = a + (size_t)b * C * HW + p;
        const float* pc = c + (size_t)b * C * HW + yy * W + xx;
        for (int cc = 0; cc < C; ++cc)
            acc += pa[cc * HW] * pc[cc * HW];
    }
    out[idx] = leaky(acc * (1.0f / (float)C));
}

// ---------------------------------------------------------------------------
// Bilinear sampling helpers (match the reference grid_sample exactly)
// ---------------------------------------------------------------------------
__device__ __forceinline__ void samp_setup(float g, int N, int& i0, int& i1, float& wf)
{
    float gp = (g + 1.0f) * ((float)N * 0.5f) - 0.5f;
    gp = fminf(fmaxf(gp, 0.0f), (float)(N - 1));
    float f0 = floorf(gp);
    wf = gp - f0;
    int i = (int)f0;
    if (i < 0) i = 0;
    if (i > N - 1) i = N - 1;
    i0 = i;
    i1 = (i + 1 < N - 1) ? (i + 1) : (N - 1);
}

// img [B,C,H,W] sampled at grid2 [B,2,H,W] (ch0 = gx, ch1 = gy, normalized)
__global__ void k_grid_sample(const float* __restrict__ img, const float* __restrict__ grid2,
                              float* __restrict__ out, int B, int C, int H, int W)
{
    const int HW = H * W;
    const int idx = blockIdx.x * blockDim.x + threadIdx.x;
    if (idx >= B * HW) return;
    const int p = idx % HW;
    const int b = idx / HW;
    const float gx = grid2[(b * 2 + 0) * HW + p];
    const float gy = grid2[(b * 2 + 1) * HW + p];
    int x0, x1, y0, y1; float wx, wy;
    samp_setup(gx, W, x0, x1, wx);
    samp_setup(gy, H, y0, y1, wy);
    const float w00 = (1.0f - wx) * (1.0f - wy), w01 = wx * (1.0f - wy);
    const float w10 = (1.0f - wx) * wy,          w11 = wx * wy;
    const float* base = img + (size_t)b * C * HW;
    for (int cc = 0; cc < C; ++cc) {
        const float* ib = base + (size_t)cc * HW;
        const float v = ib[y0 * W + x0] * w00 + ib[y0 * W + x1] * w01 +
                        ib[y1 * W + x0] * w10 + ib[y1 * W + x1] * w11;
        out[((size_t)b * C + cc) * HW + p] = v;
    }
}

// apply_offset(off) -> normalized grid; if first, write grid as flow;
// else sample lf [B,2,H,W] at the grid. out [B,2,H,W]
__global__ void k_apply_offset_sample(const float* __restrict__ off, const float* __restrict__ lf,
                                      float* __restrict__ out, int B, int H, int W, int first)
{
    const int HW = H * W;
    const int idx = blockIdx.x * blockDim.x + threadIdx.x;
    if (idx >= B * HW) return;
    const int p = idx % HW;
    const int b = idx / HW;
    const int y = p / W, x = p % W;
    const float gx = ((float)x + off[(b * 2 + 0) * HW + p]) / ((W - 1.0f) * 0.5f) - 1.0f;
    const float gy = ((float)y + off[(b * 2 + 1) * HW + p]) / ((H - 1.0f) * 0.5f) - 1.0f;
    if (first) {
        out[(b * 2 + 0) * HW + p] = gx;
        out[(b * 2 + 1) * HW + p] = gy;
        return;
    }
    int x0, x1, y0, y1; float wx, wy;
    samp_setup(gx, W, x0, x1, wx);
    samp_setup(gy, H, y0, y1, wy);
    const float w00 = (1.0f - wx) * (1.0f - wy), w01 = wx * (1.0f - wy);
    const float w10 = (1.0f - wx) * wy,          w11 = wx * wy;
    #pragma unroll
    for (int cc = 0; cc < 2; ++cc) {
        const float* ib = lf + ((size_t)b * 2 + cc) * HW;
        out[(b * 2 + cc) * HW + p] =
            ib[y0 * W + x0] * w00 + ib[y0 * W + x1] * w01 +
            ib[y1 * W + x0] * w10 + ib[y1 * W + x1] * w11;
    }
}

// half-pixel bilinear 2x upsample:  [B,C,H,W] -> [B,C,2H,2W]
__global__ void k_upsample2x(const float* __restrict__ in, float* __restrict__ out,
                             int B, int C, int H, int W)
{
    const int Ho = 2 * H, Wo = 2 * W;
    const int idx = blockIdx.x * blockDim.x + threadIdx.x;
    if (idx >= B * C * Ho * Wo) return;
    const int p  = idx % (Ho * Wo);
    const int bc = idx / (Ho * Wo);
    const int yo = p / Wo, xo = p % Wo;
    const float sy = 0.5f * (float)yo - 0.25f;
    const float sx = 0.5f * (float)xo - 0.25f;
    const float fy = floorf(sy), fx = floorf(sx);
    const float wy = sy - fy,    wx = sx - fx;
    int y0 = (int)fy, x0 = (int)fx;
    int y0c = min(max(y0, 0), H - 1), y1c = min(max(y0 + 1, 0), H - 1);
    int x0c = min(max(x0, 0), W - 1), x1c = min(max(x0 + 1, 0), W - 1);
    const float* ib = in + (size_t)bc * H * W;
    out[idx] = ib[y0c * W + x0c] * (1.0f - wx) * (1.0f - wy) +
               ib[y0c * W + x1c] * wx * (1.0f - wy) +
               ib[y1c * W + x0c] * (1.0f - wx) * wy +
               ib[y1c * W + x1c] * wx * wy;
}

// ---------------------------------------------------------------------------
// Host orchestration
// ---------------------------------------------------------------------------
extern "C" void kernel_launch(void* const* d_in, const int* in_sizes, int n_in,
                              void* d_out, int out_size, void* d_ws, size_t ws_size,
                              hipStream_t stream)
{
    const int B = 4;
    const float* x = (const float*)d_in[0];
    const float* xw[5]; const float* xc[5];
    for (int j = 0; j < 5; ++j) {
        xw[j] = (const float*)d_in[1 + 2 * j];
        xc[j] = (const float*)d_in[2 + 2 * j];
    }

    // Param flattening order: insertion (w0,b0,w1,b1,...) vs sorted (b0..b3,w0..w3).
    // main[0].w0 has 128*49*9 = 56448 elems; main[0].b0 has 128.
    const bool sorted = (n_in > 11) && (in_sizes[11] <= 128);
    auto Wp = [&](int base, int j) { return (const float*)d_in[base + (sorted ? 4 + j : 2 * j)]; };
    auto Bp = [&](int base, int j) { return (const float*)d_in[base + (sorted ? j : 2 * j + 1)]; };

    // Workspace carve (floats). Total ~28.7M floats (~115 MB) -> fits in L2.
    float* ws = (float*)d_ws;
    size_t o = 0;
    auto carve = [&](size_t n) { float* p = ws + o; o += n; return p; };
    float* bufWarp = carve(12582912);  // 4*256*12288
    float* bufCorr = carve(2408448);   // 4*49*12288
    float* bufT0   = carve(6291456);   // 4*128*12288
    float* bufT1   = carve(6291456);
    float* bufCF   = carve(98304);     // 4*2*12288
    float* bufF1   = carve(98304);
    float* bufF2   = carve(98304);
    float* lfA     = carve(393216);    // 4*2*49152
    float* lfB     = carve(393216);

    const int resH[5] = {128, 64, 32, 16, 8};
    const int resW[5] = {96, 48, 24, 12, 6};

    auto conv = [&](const float* in0, int Cin0, const float* in1, int Cin1,
                    const float* w, const float* bias, float* out_,
                    int Cout, int H, int Wd, int lk) {
        const int HW = H * Wd;
        dim3 grid((HW + 63) / 64, 1, B);
        switch ((Cout + 15) / 16) {
        case 8:
            k_conv3x3_wmma<8><<<grid, 128, 0, stream>>>(in0, Cin0, in1, Cin1, w,
                                                        bias, out_, Cout, H, Wd, lk);
            break;
        case 4:
            k_conv3x3_wmma<4><<<grid, 128, 0, stream>>>(in0, Cin0, in1, Cin1, w,
                                                        bias, out_, Cout, H, Wd, lk);
            break;
        case 2:
            k_conv3x3_wmma<2><<<grid, 128, 0, stream>>>(in0, Cin0, in1, Cin1, w,
                                                        bias, out_, Cout, H, Wd, lk);
            break;
        default:
            k_conv3x3_wmma<1><<<grid, 128, 0, stream>>>(in0, Cin0, in1, Cin1, w,
                                                        bias, out_, Cout, H, Wd, lk);
            break;
        }
    };

    float* lf = nullptr;
    const int blk = 256;

    for (int i = 0; i < 5; ++i) {
        const int lev = 4 - i;                       // coarse -> fine
        const int H = resH[lev], Wd = resW[lev], HW = H * Wd;
        const float* xwp = xw[lev];
        const float* xcp = xc[lev];
        const int nPix = B * HW;

        // 1. warp x_warp by last_flow (skip at coarsest level)
        const float* firstPtr = xwp;
        if (lf) {
            k_grid_sample<<<(nPix + blk - 1) / blk, blk, 0, stream>>>(
                xwp, lf, bufWarp, B, 256, H, Wd);
            firstPtr = bufWarp;
        }
        // 2. correlation + leaky
        {
            const int tot = B * 49 * HW;
            k_corr7<<<(tot + blk - 1) / blk, blk, 0, stream>>>(
                firstPtr, xcp, bufCorr, B, 256, H, Wd);
        }
        // 3. main conv stack: 49 -> 128 -> 64 -> 32 -> 2
        {
            const int bm = 11 + i * 8;
            conv(bufCorr, 49,  nullptr, 0, Wp(bm, 0), Bp(bm, 0), bufT0, 128, H, Wd, 1);
            conv(bufT0,   128, nullptr, 0, Wp(bm, 1), Bp(bm, 1), bufT1, 64,  H, Wd, 1);
            conv(bufT1,   64,  nullptr, 0, Wp(bm, 2), Bp(bm, 2), bufT0, 32,  H, Wd, 1);
            conv(bufT0,   32,  nullptr, 0, Wp(bm, 3), Bp(bm, 3), bufCF, 2,   H, Wd, 0);
        }
        // 4. flow = (first ? grid : grid_sample(last_flow, grid))
        k_apply_offset_sample<<<(nPix + blk - 1) / blk, blk, 0, stream>>>(
            bufCF, lf, bufF1, B, H, Wd, lf ? 0 : 1);
        // 5. xw = grid_sample(x_warp, flow)
        k_grid_sample<<<(nPix + blk - 1) / blk, blk, 0, stream>>>(
            xwp, bufF1, bufWarp, B, 256, H, Wd);
        // 6. refine conv stack on concat(xw, x_cond): (256+256) -> 128 -> 64 -> 32 -> 2
        {
            const int br = 51 + i * 8;
            conv(bufWarp, 256, xcp, 256, Wp(br, 0), Bp(br, 0), bufT0, 128, H, Wd, 1);
            conv(bufT0,   128, nullptr, 0, Wp(br, 1), Bp(br, 1), bufT1, 64, H, Wd, 1);
            conv(bufT1,   64,  nullptr, 0, Wp(br, 2), Bp(br, 2), bufT0, 32, H, Wd, 1);
            conv(bufT0,   32,  nullptr, 0, Wp(br, 3), Bp(br, 3), bufCF, 2,  H, Wd, 0);
        }
        // 7. flow2 = grid_sample(flow, apply_offset(refine_flow))
        k_apply_offset_sample<<<(nPix + blk - 1) / blk, blk, 0, stream>>>(
            bufCF, bufF1, bufF2, B, H, Wd, 0);
        // 8. last_flow = upsample2x(flow2)
        float* nlf = (lf == lfA) ? lfB : lfA;
        const int tot = B * 2 * (2 * H) * (2 * Wd);
        k_upsample2x<<<(tot + blk - 1) / blk, blk, 0, stream>>>(bufF2, nlf, B, 2, H, Wd);
        lf = nlf;
    }

    // final: x_warp = grid_sample(x, last_flow); outputs = [x_warp | last_flow]
    {
        const int H = 256, Wd = 192;
        const int nPix = B * H * Wd;
        k_grid_sample<<<(nPix + blk - 1) / blk, blk, 0, stream>>>(
            x, lf, (float*)d_out, B, 3, H, Wd);
        hipMemcpyAsync((float*)d_out + (size_t)B * 3 * H * Wd, lf,
                       (size_t)B * 2 * H * Wd * sizeof(float),
                       hipMemcpyDeviceToDevice, stream);
    }
}
